// Backbone_20332375179599
// MI455X (gfx1250) — 8-bit matrix operand order repaired and checked by reading, compiled, not run on hardware
//
#include <hip/hip_runtime.h>


typedef __attribute__((ext_vector_type(16))) int   v16i;
typedef __attribute__((ext_vector_type(8)))  float v8f;

#define L_TOK   512
#define N_NODE  288
#define KDIM    576
#define KPAD    640
#define PAD_D   100.0f
#define FP8_ONE 0x38

__global__ __launch_bounds__(128, 1)
void pack_rows_kernel(const float* __restrict__ top,
                      const float* __restrict__ left,
                      unsigned char* __restrict__ X,
                      float* __restrict__ rowsum)
{
    const int row = blockIdx.x;
    const float* t = top  + (size_t)row * N_NODE;
    const float* l = left + (size_t)row * N_NODE;
    unsigned char* x = X  + (size_t)row * KPAD;

    float s = 0.0f;
    for (int n = threadIdx.x; n < N_NODE; n += 128) {
        float tv = t[n], lv = l[n];
        x[n]          = (tv != 0.0f) ? (unsigned char)FP8_ONE : 0;
        x[N_NODE + n] = (lv != 0.0f) ? (unsigned char)FP8_ONE : 0;
        s += tv + lv;
    }
    if (threadIdx.x < KPAD - KDIM) x[KDIM + threadIdx.x] = 0;

    __shared__ float sm[128];
    sm[threadIdx.x] = s;
    __syncthreads();
    for (int off = 64; off > 0; off >>= 1) {
        if (threadIdx.x < off) sm[threadIdx.x] += sm[threadIdx.x + off];
        __syncthreads();
    }
    if (threadIdx.x == 0) rowsum[row] = sm[0];
}

__global__ __launch_bounds__(128, 1)
void gram_attn_kernel(const unsigned char* __restrict__ X,
                      const float* __restrict__ rowsum,
                      const int* __restrict__ ind,
                      float* __restrict__ out)
{
    const int b    = blockIdx.z;
    const int wave = threadIdx.x >> 5;
    const int lane = threadIdx.x & 31;
    const int rowBlk = blockIdx.y * 64 + (wave >> 1) * 32;
    const int colBlk = blockIdx.x * 64 + (wave & 1)  * 32;

    const unsigned char* Xb = X + (size_t)b * L_TOK * KPAD;
    const int lm  = lane & 15;
    const int hiH = lane >> 4;

    const unsigned char* aR[2]; const unsigned char* bR[2];
    aR[0] = Xb + (size_t)(rowBlk + lm) * KPAD;
    aR[1] = aR[0] + (size_t)16 * KPAD;
    bR[0] = Xb + (size_t)(colBlk + lm) * KPAD;
    bR[1] = bR[0] + (size_t)16 * KPAD;

    v8f acc[2][2] = {};

    #pragma unroll
    for (int k = 0; k < KPAD; k += 128) {
        v16i a[2], bf[2];
        #pragma unroll
        for (int t = 0; t < 2; ++t) {
            union { v16i v; uint2 d[8]; } ua;
            #pragma unroll
            for (int h = 0; h < 2; ++h)
                #pragma unroll
                for (int s = 0; s < 4; ++s)
                    ua.d[h * 4 + s] =
                        *(const uint2*)(aR[t] + k + h * 64 + (s >> 1) * 32 + hiH * 16 + (s & 1) * 8);
            a[t] = ua.v;
            union { v16i v; uint4 q4[4]; } ub;
            #pragma unroll
            for (int q = 0; q < 4; ++q)
                ub.q4[q] = *(const uint4*)(bR[t] + k + q * 32 + hiH * 16);
            bf[t] = ub.v;
        }
        #pragma unroll
        for (int ti = 0; ti < 2; ++ti)
            #pragma unroll
            for (int tj = 0; tj < 2; ++tj)
                acc[ti][tj] = __builtin_amdgcn_wmma_f32_16x16x128_fp8_fp8(
                    a[ti], bf[tj], (short)0, acc[ti][tj], false, false);
    }

    const int bL = b * L_TOK;
    float ri[2][8]; int ii[2][8];
    #pragma unroll
    for (int ti = 0; ti < 2; ++ti)
        #pragma unroll
        for (int v = 0; v < 8; ++v) {
            const int i = rowBlk + ti * 16 + hiH * 8 + v;
            ri[ti][v] = rowsum[bL + i];
            ii[ti][v] = ind[bL + i];
        }
    float rj[2]; int ij[2];
    #pragma unroll
    for (int tj = 0; tj < 2; ++tj) {
        const int j = colBlk + tj * 16 + lm;
        rj[tj] = rowsum[bL + j];
        ij[tj] = ind[bL + j];
    }

    #pragma unroll
    for (int ti = 0; ti < 2; ++ti)
        #pragma unroll
        for (int tj = 0; tj < 2; ++tj) {
            const int jg = colBlk + tj * 16 + lm;
            #pragma unroll
            for (int v = 0; v < 8; ++v) {
                const int ig = rowBlk + ti * 16 + hiH * 8 + v;
                float raw = ri[ti][v] + rj[tj] - 2.0f * acc[ti][tj][v];
                const int ia = ii[ti][v], ja = ij[tj];
                const bool cls = (ia == -1) || (ja == -1);
                const bool pad = (ia == 0)  || (ja == 0);
                const bool sep = (ia > 0) && (ia & 1) && (ja > 0) && (ja & 1);
                float val = (cls ? 0.0f : raw) + (pad ? PAD_D : 0.0f);
                val *= sep ? 1.0f : (1.0f + PAD_D);
                out[((size_t)(bL + ig)) * L_TOK + jg] = val;
            }
        }
}

extern "C" void kernel_launch(void* const* d_in, const int* in_sizes, int n_in,
                              void* d_out, int out_size, void* d_ws, size_t ws_size,
                              hipStream_t stream) {
    (void)in_sizes; (void)n_in; (void)out_size; (void)ws_size;
    const float* top  = (const float*)d_in[0];
    const float* left = (const float*)d_in[1];
    const int*   ind  = (const int*)d_in[2];
    float* out = (float*)d_out;

    unsigned char* X      = (unsigned char*)d_ws;
    float*         rowsum = (float*)((char*)d_ws + (size_t)2 * L_TOK * KPAD);

    pack_rows_kernel<<<dim3(2 * L_TOK), dim3(128), 0, stream>>>(top, left, X, rowsum);
    gram_attn_kernel<<<dim3(L_TOK / 64, L_TOK / 64, 2), dim3(128), 0, stream>>>(
        X, rowsum, ind, out);
}
